// Attention_47545287967487
// MI455X (gfx1250) — matrix operand order repaired and checked by reading, not compiled, not run on hardware
//
#include <hip/hip_runtime.h>
#include <hip/hip_bf16.h>
#include <math.h>

// ---------------------------------------------------------------------------
// Problem constants (reference: B=2, S=2048, D=1024, DH=64, H=16)
// ---------------------------------------------------------------------------
#define BB   2
#define SS   2048
#define DD   1024
#define DH   64
#define HH   16
#define MM   (BB * SS)          // 4096 rows
#define EPS  1e-6f
#define SM_SCALE 0.125f         // 1/sqrt(64)

typedef __attribute__((ext_vector_type(16))) __bf16 v16bf;
typedef __attribute__((ext_vector_type(8)))  float  v8f;

#define DEV __device__ __forceinline__

DEV unsigned short f2bf(float f) {
    unsigned int u = __float_as_uint(f);
    unsigned int r = u + 0x7FFFu + ((u >> 16) & 1u);   // RNE
    return (unsigned short)(r >> 16);
}

union F16x16 { v16bf v; uint4 q[2]; unsigned short u[16]; };

DEV v8f vzero8() {
    v8f v;
    #pragma unroll
    for (int i = 0; i < 8; ++i) v[i] = 0.f;
    return v;
}

// ---------------------------------------------------------------------------
// f32 -> bf16 elementwise
// ---------------------------------------------------------------------------
__global__ __launch_bounds__(256)
void cvt_bf16_kernel(const float* __restrict__ X, unsigned short* __restrict__ Y, int n) {
    int i = blockIdx.x * 256 + threadIdx.x;
    if (i < n) Y[i] = f2bf(X[i]);
}

// ---------------------------------------------------------------------------
// W[K][N] f32 -> WT[N][K] bf16 (transpose + convert)
// ---------------------------------------------------------------------------
__global__ __launch_bounds__(256)
void wtrans_kernel(const float* __restrict__ W, unsigned short* __restrict__ WT) {
    int i = blockIdx.x * 256 + threadIdx.x;       // output index, N-major
    int n = i >> 10;
    int k = i & 1023;
    WT[i] = f2bf(W[k * DD + n]);
}

// ---------------------------------------------------------------------------
// GEMM: C[M,N] (f32) = A[M,K] (bf16, row-major) * WT[N,K] (bf16, K contiguous)
// Block tile 256x128, 8 waves (4x2), wave tile 64x64 (16 WMMA / K-step),
// K-step 32. ds:wmma ratio 1.0 -> LDS bandwidth no longer the limiter.
// ---------------------------------------------------------------------------
__global__ __launch_bounds__(256)
void gemm_bf16_kernel(const unsigned short* __restrict__ A,
                      const unsigned short* __restrict__ WT,
                      float* __restrict__ C, int M, int N, int K) {
    __shared__ __align__(16) unsigned short As[256 * 32];   // 16 KB
    __shared__ __align__(16) unsigned short Bs[128 * 32];   //  8 KB

    const int tid  = threadIdx.x;
    const int lane = tid & 31;
    const int w    = tid >> 5;         // 0..7
    const int wm   = w >> 1;           // 0..3  (M direction, 64 rows each)
    const int wn   = w & 1;            // 0..1  (N direction, 64 cols each)
    const int m0   = blockIdx.y * 256;
    const int n0   = blockIdx.x * 128;
    const int nl   = lane & 15;
    const int hi   = lane >> 4;        // 0/1 lane half

    const int brow  = tid >> 1;        // B staging row 0..127
    const int bhalf = tid & 1;         // B staging 16-elem half

    v8f acc[4][4];
    #pragma unroll
    for (int i = 0; i < 4; ++i)
        #pragma unroll
        for (int j = 0; j < 4; ++j) acc[i][j] = vzero8();

    for (int k0 = 0; k0 < K; k0 += 32) {
        // Stage A tile 256x32 (1 row / thread) and B tile 128x32
        {
            const uint4* ga = reinterpret_cast<const uint4*>(A + (size_t)(m0 + tid) * K + k0);
            uint4* sa = reinterpret_cast<uint4*>(As + tid * 32);
            sa[0] = ga[0]; sa[1] = ga[1];
            const uint4* gb = reinterpret_cast<const uint4*>(WT + (size_t)(n0 + brow) * K + k0 + bhalf * 16);
            uint4* sb = reinterpret_cast<uint4*>(Bs + brow * 32 + bhalf * 16);
            sb[0] = gb[0];
        }
        if (k0 + 32 < K) {
            __builtin_prefetch(A  + (size_t)(m0 + tid) * K + k0 + 32, 0, 1);
            __builtin_prefetch(WT + (size_t)(n0 + brow) * K + k0 + 32, 0, 1);
        }
        __syncthreads();

        // A fragments: 16-bit A layout, two contiguous 8-elem chunks per lane
        F16x16 a[4], b[4];
        #pragma unroll
        for (int i = 0; i < 4; ++i) {
            const uint4* r = reinterpret_cast<const uint4*>(As + (wm * 64 + i * 16 + nl) * 32);
            a[i].q[0] = r[hi];          // K base {0,8}
            a[i].q[1] = r[hi + 2];      // K base {16,24}
        }
        // B fragments: lane = column N, two 8-elem K chunks per lane (same order as A)
        #pragma unroll
        for (int j = 0; j < 4; ++j) {
            const uint4* r = reinterpret_cast<const uint4*>(Bs + (wn * 64 + j * 16 + nl) * 32);
            b[j].q[0] = r[hi];
            b[j].q[1] = r[hi + 2];
        }
        #pragma unroll
        for (int i = 0; i < 4; ++i)
            #pragma unroll
            for (int j = 0; j < 4; ++j)
                acc[i][j] = __builtin_amdgcn_wmma_f32_16x16x32_bf16(
                    false, a[i].v, false, b[j].v, (short)0, acc[i][j], false, false);
        __syncthreads();
    }

    // Epilogue: C layout -> lane nl = col, VGPR r = row (+8 for hi half)
    #pragma unroll
    for (int i = 0; i < 4; ++i)
        #pragma unroll
        for (int j = 0; j < 4; ++j)
            #pragma unroll
            for (int r = 0; r < 8; ++r) {
                int row = m0 + wm * 64 + i * 16 + r + hi * 8;
                int col = n0 + wn * 64 + j * 16 + nl;
                C[(size_t)row * N + col] = acc[i][j][r];
            }
}

// ---------------------------------------------------------------------------
// RMS-norm per (token, head) over DH=64 (or plain convert when norm==0),
// permuting [B,S,H,DH] f32 -> [B,H,S,DH] bf16. One wave32 per group.
// ---------------------------------------------------------------------------
__global__ __launch_bounds__(256)
void rms_permute_kernel(const float* __restrict__ X, unsigned short* __restrict__ Y, int norm) {
    const int warp = threadIdx.x >> 5;
    const int lane = threadIdx.x & 31;
    const int gid  = blockIdx.x * 8 + warp;          // (b*S + s)*H + h
    const int h    = gid & (HH - 1);
    const int s    = (gid / HH) & (SS - 1);
    const int b    = gid / (SS * HH);

    const float2 v = *reinterpret_cast<const float2*>(X + (size_t)gid * DH + lane * 2);
    float ss = v.x * v.x + v.y * v.y;
    #pragma unroll
    for (int m = 1; m < 32; m <<= 1) ss += __shfl_xor(ss, m, 32);
    const float sc = norm ? rsqrtf(ss * (1.f / DH) + EPS) : 1.f;

    unsigned int packed = (unsigned int)f2bf(v.x * sc) | ((unsigned int)f2bf(v.y * sc) << 16);
    size_t idx = ((size_t)(b * HH + h) * SS + s) * DH + lane * 2;
    *reinterpret_cast<unsigned int*>(Y + idx) = packed;
}

// ---------------------------------------------------------------------------
// Flash-style causal attention per (b, h, 64-row q tile). 4 waves, each
// owns 16 q-rows. QK^T and P@V via bf16 WMMA; online softmax in f32.
// ---------------------------------------------------------------------------
__global__ __launch_bounds__(128)
void attn_kernel(const unsigned short* __restrict__ Qn,
                 const unsigned short* __restrict__ Kn,
                 const unsigned short* __restrict__ Vn,
                 unsigned short* __restrict__ Ob) {
    __shared__ __align__(16) unsigned short Qs[64 * 64];
    __shared__ __align__(16) unsigned short Ks[64 * 64];
    __shared__ __align__(16) unsigned short Vt[64 * 72];   // [d][key], padded
    __shared__ __align__(16) unsigned short Ps[4 * 16 * 64];

    const int tid  = threadIdx.x;
    const int lane = tid & 31;
    const int w    = tid >> 5;             // 0..3
    const int nl   = lane & 15;
    const int hi   = lane >> 4;
    const int q0   = blockIdx.x * 64;
    const int h    = blockIdx.y;
    const int b    = blockIdx.z;
    const size_t headBase = (size_t)(b * HH + h) * SS * DH;

    // Load Q tile [64 q][64 d] row-major
    {
        int row = tid >> 1, hf = tid & 1;
        const uint4* g = reinterpret_cast<const uint4*>(Qn + headBase + (size_t)(q0 + row) * DH + hf * 32);
        uint4* s = reinterpret_cast<uint4*>(Qs + row * 64 + hf * 32);
        s[0] = g[0]; s[1] = g[1]; s[2] = g[2]; s[3] = g[3];
    }

    float mrow[8], lrow[8];
    #pragma unroll
    for (int r = 0; r < 8; ++r) { mrow[r] = -1.0e30f; lrow[r] = 0.f; }
    v8f o[4];
    #pragma unroll
    for (int j = 0; j < 4; ++j) o[j] = vzero8();

    unsigned short* pw = Ps + w * 16 * 64;

    for (int t0 = 0; t0 <= q0; t0 += 64) {
        __syncthreads();   // previous-iteration consumers done (also covers Q load, iter 0)

        // K tile [64 key][64 d] row-major (== B layout for QK^T)
        {
            int row = tid >> 1, hf = tid & 1;
            const uint4* g = reinterpret_cast<const uint4*>(Kn + headBase + (size_t)(t0 + row) * DH + hf * 32);
            uint4* s = reinterpret_cast<uint4*>(Ks + row * 64 + hf * 32);
            s[0] = g[0]; s[1] = g[1]; s[2] = g[2]; s[3] = g[3];
        }
        // V tile staged transposed: Vt[d][key] (== B layout for P@V)
        {
            int key = tid >> 1, dh = tid & 1;
            union { uint4 q[4]; unsigned short u[32]; } tmp;
            const uint4* g = reinterpret_cast<const uint4*>(Vn + headBase + (size_t)(t0 + key) * DH + dh * 32);
            tmp.q[0] = g[0]; tmp.q[1] = g[1]; tmp.q[2] = g[2]; tmp.q[3] = g[3];
            #pragma unroll
            for (int i = 0; i < 32; ++i) Vt[(dh * 32 + i) * 72 + key] = tmp.u[i];
        }
        __syncthreads();

        // Scores S = Q K^T  (16 q rows x 64 keys per wave)
        v8f s4[4];
        #pragma unroll
        for (int j = 0; j < 4; ++j) s4[j] = vzero8();
        #pragma unroll
        for (int ks = 0; ks < 64; ks += 32) {
            F16x16 a;
            const uint4* ra = reinterpret_cast<const uint4*>(Qs + (w * 16 + nl) * 64);
            a.q[0] = ra[ks / 8 + hi];
            a.q[1] = ra[ks / 8 + hi + 2];
            #pragma unroll
            for (int j = 0; j < 4; ++j) {
                F16x16 bb;
                const uint4* rb = reinterpret_cast<const uint4*>(Ks + (j * 16 + nl) * 64);
                bb.q[0] = rb[ks / 8 + hi];
                bb.q[1] = rb[ks / 8 + hi + 2];
                s4[j] = __builtin_amdgcn_wmma_f32_16x16x32_bf16(
                    false, a.v, false, bb.v, (short)0, s4[j], false, false);
            }
        }

        // Scale + causal mask (exact: key_global > q_global -> -inf)
        const int qbase = q0 + w * 16 + hi * 8;
        #pragma unroll
        for (int j = 0; j < 4; ++j)
            #pragma unroll
            for (int r = 0; r < 8; ++r) {
                float sv = s4[j][r] * SM_SCALE;
                if (t0 + j * 16 + nl > qbase + r) sv = -1.0e30f;
                s4[j][r] = sv;
            }

        // Online softmax: row max / correction / exp / row sum
        float mnew[8], cf[8], psum[8];
        #pragma unroll
        for (int r = 0; r < 8; ++r) {
            float mx = s4[0][r];
            #pragma unroll
            for (int j = 1; j < 4; ++j) mx = fmaxf(mx, s4[j][r]);
            #pragma unroll
            for (int m = 1; m < 16; m <<= 1) mx = fmaxf(mx, __shfl_xor(mx, m, 32));
            mnew[r] = fmaxf(mrow[r], mx);
            cf[r]   = __expf(mrow[r] - mnew[r]);
            mrow[r] = mnew[r];
            psum[r] = 0.f;
        }
        #pragma unroll
        for (int j = 0; j < 4; ++j)
            #pragma unroll
            for (int r = 0; r < 8; ++r) {
                float p = __expf(s4[j][r] - mnew[r]);
                psum[r] += p;
                pw[(r + hi * 8) * 64 + j * 16 + nl] = f2bf(p);   // wave-private buffer
            }
        #pragma unroll
        for (int r = 0; r < 8; ++r) {
            float ps = psum[r];
            #pragma unroll
            for (int m = 1; m < 16; m <<= 1) ps += __shfl_xor(ps, m, 32);
            lrow[r] = lrow[r] * cf[r] + ps;
        }
        #pragma unroll
        for (int j = 0; j < 4; ++j)
            #pragma unroll
            for (int r = 0; r < 8; ++r) o[j][r] = o[j][r] * cf[r];

        // O += P @ V   (P from wave-private LDS; same-wave LDS ops are in order)
        #pragma unroll
        for (int ks = 0; ks < 64; ks += 32) {
            F16x16 a;
            const uint4* ra = reinterpret_cast<const uint4*>(pw + nl * 64);
            a.q[0] = ra[ks / 8 + hi];
            a.q[1] = ra[ks / 8 + hi + 2];
            #pragma unroll
            for (int j = 0; j < 4; ++j) {
                F16x16 bb;
                const uint4* rb = reinterpret_cast<const uint4*>(Vt + (j * 16 + nl) * 72);
                bb.q[0] = rb[ks / 8 + hi];
                bb.q[1] = rb[ks / 8 + hi + 2];
                o[j] = __builtin_amdgcn_wmma_f32_16x16x32_bf16(
                    false, a.v, false, bb.v, (short)0, o[j], false, false);
            }
        }
    }

    // Normalize and store O as bf16 [B,S,D] (ready to be GEMM-A for Wo)
    #pragma unroll
    for (int j = 0; j < 4; ++j)
        #pragma unroll
        for (int r = 0; r < 8; ++r) {
            float ov = o[j][r] / lrow[r];
            int srow = q0 + w * 16 + r + hi * 8;
            size_t idx = ((size_t)(b * SS + srow)) * DD + h * DH + j * 16 + nl;
            Ob[idx] = f2bf(ov);
        }
}

// ---------------------------------------------------------------------------
// Host-side orchestration
// ---------------------------------------------------------------------------
extern "C" void kernel_launch(void* const* d_in, const int* in_sizes, int n_in,
                              void* d_out, int out_size, void* d_ws, size_t ws_size,
                              hipStream_t stream) {
    const float* x  = (const float*)d_in[0];
    const float* Wq = (const float*)d_in[1];
    const float* Wk = (const float*)d_in[2];
    const float* Wv = (const float*)d_in[3];
    const float* Wo = (const float*)d_in[4];
    // d_in[5] = causal mask: computed analytically in-kernel, ignored.

    char* ws = (char*)d_ws;
    const size_t SZ_XB = (size_t)MM * DD * 2;       // 8 MB  bf16 x
    const size_t SZ_WT = (size_t)DD * DD * 2;       // 2 MB  bf16 W^T
    const size_t SZ_CF = (size_t)MM * DD * 4;       // 16 MB f32 GEMM scratch
    const size_t SZ_HN = (size_t)MM * DD * 2;       // 8 MB  bf16 head-major

    unsigned short* xb  = (unsigned short*)(ws);                     size_t off = SZ_XB;
    unsigned short* WqT = (unsigned short*)(ws + off); off += SZ_WT;
    unsigned short* WkT = (unsigned short*)(ws + off); off += SZ_WT;
    unsigned short* WvT = (unsigned short*)(ws + off); off += SZ_WT;
    unsigned short* WoT = (unsigned short*)(ws + off); off += SZ_WT;
    float*          Cf  = (float*)(ws + off);          off += SZ_CF;
    unsigned short* Qn  = (unsigned short*)(ws + off); off += SZ_HN;
    unsigned short* Kn  = (unsigned short*)(ws + off); off += SZ_HN;
    unsigned short* Vn  = (unsigned short*)(ws + off); off += SZ_HN;
    unsigned short* Ob  = (unsigned short*)(ws + off); off += SZ_HN;

    // 1) Precision conversion / weight transposition
    cvt_bf16_kernel<<<(MM * DD) / 256, 256, 0, stream>>>(x, xb, MM * DD);
    wtrans_kernel<<<(DD * DD) / 256, 256, 0, stream>>>(Wq, WqT);
    wtrans_kernel<<<(DD * DD) / 256, 256, 0, stream>>>(Wk, WkT);
    wtrans_kernel<<<(DD * DD) / 256, 256, 0, stream>>>(Wv, WvT);
    wtrans_kernel<<<(DD * DD) / 256, 256, 0, stream>>>(Wo, WoT);

    dim3 ggrid(DD / 128, MM / 256);   // (8, 16)

    // 2) Q = x@Wq, RMS-norm, permute to head-major bf16
    gemm_bf16_kernel<<<ggrid, 256, 0, stream>>>(xb, WqT, Cf, MM, DD, DD);
    rms_permute_kernel<<<(MM * HH) / 8, 256, 0, stream>>>(Cf, Qn, 1);
    // 3) K likewise
    gemm_bf16_kernel<<<ggrid, 256, 0, stream>>>(xb, WkT, Cf, MM, DD, DD);
    rms_permute_kernel<<<(MM * HH) / 8, 256, 0, stream>>>(Cf, Kn, 1);
    // 4) V: no norm, just permute/convert
    gemm_bf16_kernel<<<ggrid, 256, 0, stream>>>(xb, WvT, Cf, MM, DD, DD);
    rms_permute_kernel<<<(MM * HH) / 8, 256, 0, stream>>>(Cf, Vn, 0);

    // 5) Causal flash attention
    attn_kernel<<<dim3(SS / 64, HH, BB), 128, 0, stream>>>(Qn, Kn, Vn, Ob);

    // 6) Output projection straight into d_out (f32)
    gemm_bf16_kernel<<<ggrid, 256, 0, stream>>>(Ob, WoT, (float*)d_out, MM, DD, DD);
}
